// Net_29025388986629
// MI455X (gfx1250) — matrix operand order repaired and checked by reading, not compiled, not run on hardware
//
#include <hip/hip_runtime.h>

// GCN forward: 1 input conv + 19 hidden convs + final FC.
// GEMMs: v_wmma_f32_16x16x32_bf16 (bf16 in, f32 acc); activations kept bf16 in
// global memory so LDS staging is pure B128 copies (async-to-LDS, ASYNCcnt).
// Scatter-add uses global_atomic_add_f32; working set is L2-resident (192MB).

#define NN   50000   // nodes
#define NE   600000  // edges
#define INF  16      // input feats
#define HF   128     // hidden feats
#define NC   4       // classes
#define NL   19      // hidden layers

typedef __attribute__((ext_vector_type(8)))  float          v8f;
typedef __attribute__((ext_vector_type(8)))  __bf16         v8bf;
typedef __attribute__((ext_vector_type(16))) __bf16         v16bf;
typedef __attribute__((ext_vector_type(8)))  unsigned short v8u16;
typedef int v4i __attribute__((vector_size(16)));  // matches builtin's V4i

#if __has_builtin(__builtin_amdgcn_global_load_async_to_lds_b128)
#define HAVE_ASYNC 1
#else
#define HAVE_ASYNC 0
#endif

#if __has_builtin(__builtin_amdgcn_s_wait_asynccnt)
#define WAIT_ASYNC() __builtin_amdgcn_s_wait_asynccnt(0)
#else
#define WAIT_ASYNC() asm volatile("s_wait_asynccnt 0" ::: "memory")
#endif

__device__ __forceinline__ unsigned short f2bf(float f) {
  // round-to-nearest-even f32 -> bf16
  unsigned int u = __float_as_uint(f);
  unsigned int r = u + 0x7FFFu + ((u >> 16) & 1u);
  return (unsigned short)(r >> 16);
}

__device__ __forceinline__ float bf2f(unsigned short u) {
  return __uint_as_float(((unsigned int)u) << 16);
}

// 16-byte global -> LDS copy (async, tracked by ASYNCcnt)
__device__ __forceinline__ void copy16(const unsigned short* g, unsigned short* l) {
#if HAVE_ASYNC
  __builtin_amdgcn_global_load_async_to_lds_b128(
      (__attribute__((address_space(1))) v4i*)g,
      (__attribute__((address_space(3))) v4i*)l, 0, 0);
#else
  *reinterpret_cast<v8u16*>(l) = *reinterpret_cast<const v8u16*>(g);
#endif
}

__device__ __forceinline__ v16bf frag16(const unsigned short* lo,
                                        const unsigned short* hi) {
  v8bf a = *reinterpret_cast<const v8bf*>(lo);   // ds_load_b128
  v8bf b = *reinterpret_cast<const v8bf*>(hi);   // ds_load_b128
  v16bf r;
#pragma unroll
  for (int i = 0; i < 8; ++i) { r[i] = a[i]; r[i + 8] = b[i]; }
  return r;
}

// ---------------------------------------------------------------- utilities
__global__ void k_fill0(float* __restrict__ p, int n) {
  int i = blockIdx.x * blockDim.x + threadIdx.x;
  if (i < n) p[i] = 0.0f;
}

__global__ void k_deg(const int* __restrict__ ei, float* __restrict__ deg) {
  int e = blockIdx.x * blockDim.x + threadIdx.x;
  if (e < NE) atomicAdd(&deg[ei[NE + e]], 1.0f);
}

__global__ void k_dinv(const float* __restrict__ deg, float* __restrict__ dinv) {
  int i = blockIdx.x * blockDim.x + threadIdx.x;
  if (i < NN) dinv[i] = 1.0f / fmaxf(deg[i], 1.0f);
}

// W_h [l][k][n] f32  ->  Wbf [l][n][k] bf16 (pre-transposed for B-fragments)
__global__ void k_cvt_bf16t(const float* __restrict__ w,
                            unsigned short* __restrict__ o) {
  int i = blockIdx.x * blockDim.x + threadIdx.x;
  if (i < NL * HF * HF) {
    int l = i >> 14;
    int r = i & 16383;
    int n = r >> 7, k = r & 127;
    o[i] = f2bf(w[(l << 14) + k * HF + n]);
  }
}

// ------------------------------------------------- layer 0: t = x@W_in + b_in
__global__ void k_gemm_in(const float* __restrict__ x, const float* __restrict__ W,
                          const float* __restrict__ b, float* __restrict__ t) {
  __shared__ float xs[INF];
  int node = blockIdx.x;
  int j = threadIdx.x;                    // 0..127 output feature
  if (j < INF) xs[j] = x[(size_t)node * INF + j];
  __syncthreads();
  float acc = b[j];
#pragma unroll
  for (int k = 0; k < INF; ++k) acc += xs[k] * W[k * HF + j];
  t[(size_t)node * HF + j] = acc;
}

// ----------------------------- hidden GEMM: t = h@W + b  (WMMA bf16, f32 acc)
__global__ __launch_bounds__(256)
void k_gemm_wmma(const unsigned short* __restrict__ hbf,  // bf16 [m][k]
                 const unsigned short* __restrict__ Wg,   // bf16 [n][k] (pre-T)
                 const float* __restrict__ bias,
                 float* __restrict__ tout, int nNodes) {
  // W in LDS: [n][k], stride 136 halves (272B = 17x16B, bank-spread)
  __shared__ alignas(16) unsigned short W_lds[HF * 136];
  // A chunk in LDS: [m][k'], 32-deep K slice, stride 40 halves (80B)
  __shared__ alignas(16) unsigned short A_lds[128 * 40];

  const int tid   = threadIdx.x;
  const int wave  = tid >> 5;        // 0..7 -> N-tile
  const int lane  = tid & 31;
  const int lrow  = lane & 15;       // row/col within tile
  const int lhi   = lane >> 4;       // half-wave select
  const int mBase = blockIdx.x * 128;
  const int n     = wave * 16 + lrow;

  // stage full W: 2048 x 16B straight copies (coalesced, async)
#pragma unroll
  for (int c = tid; c < HF * HF / 8; c += 256) {
    int nn = c >> 4;
    int ko = (c & 15) * 8;
    copy16(&Wg[nn * HF + ko], &W_lds[nn * 136 + ko]);
  }

  v8f acc[8] = {};   // 8 M-tiles x (16x16 f32) per wave

  for (int kc = 0; kc < HF; kc += 32) {
    __syncthreads();   // prev-chunk reads done (covers W staging on iter 0)
    // stage 128 rows x 32 K of bf16 activations: 512 x 16B copies
#pragma unroll
    for (int c = tid; c < 512; c += 256) {
      int m  = c >> 2;
      int ko = (c & 3) * 8;
      int row = mBase + m;
      if (row >= nNodes) row = nNodes - 1;   // clamp: rows masked at store
      copy16(&hbf[(size_t)row * HF + kc + ko], &A_lds[m * 40 + ko]);
    }
    WAIT_ASYNC();
    __syncthreads();

    // B fragment (K x N = 32x16): lanes 0-15 hold K=kc..kc+15, 16-31 hold +16
    v16bf bfrag = frag16(&W_lds[n * 136 + kc + lhi * 16],
                         &W_lds[n * 136 + kc + lhi * 16 + 8]);
#pragma unroll
    for (int mt = 0; mt < 8; ++mt) {
      // A fragment (M x K = 16x32): contiguous K, lanes 0-15 hold K=0..15, 16-31 hold +16
      const unsigned short* arow = &A_lds[(mt * 16 + lrow) * 40];
      v16bf afrag = frag16(arow + lhi * 16, arow + lhi * 16 + 8);
      acc[mt] = __builtin_amdgcn_wmma_f32_16x16x32_bf16(
          false, afrag, false, bfrag, (short)0, acc[mt], false, false);
    }
  }

  // epilogue: + bias, store f32 (lanes 0-15 contiguous in N)
  float bv = bias[n];
#pragma unroll
  for (int mt = 0; mt < 8; ++mt) {
#pragma unroll
    for (int r = 0; r < 8; ++r) {
      int m = mBase + mt * 16 + lhi * 8 + r;
      if (m < nNodes) tout[(size_t)m * HF + n] = acc[mt][r] + bv;
    }
  }
}

// ------------------------- scatter: agg[dst] += t[src] * ew   (1 edge / wave)
__global__ void k_scatter(const float* __restrict__ t, const int* __restrict__ ei,
                          const float* __restrict__ ew, float* __restrict__ agg) {
  int gid  = blockIdx.x * blockDim.x + threadIdx.x;
  int e    = gid >> 5;
  int lane = gid & 31;
  if (e >= NE) return;
  int src = ei[e];
  int dst = ei[NE + e];
  float w = ew[e];
  const float4 v = *reinterpret_cast<const float4*>(t + (size_t)src * HF + lane * 4);
  float* out = agg + (size_t)dst * HF + lane * 4;
  atomicAdd(out + 0, v.x * w);
  atomicAdd(out + 1, v.y * w);
  atomicAdd(out + 2, v.z * w);
  atomicAdd(out + 3, v.w * w);
}

// ------------- h(bf16) = leaky_relu(agg * deg_inv), slope 0.01, packed stores
__global__ void k_pointwise(const float* __restrict__ agg,
                            const float* __restrict__ dinv,
                            unsigned short* __restrict__ hbf) {
  int i = blockIdx.x * blockDim.x + threadIdx.x;   // over NN*32 (4 feats each)
  if (i >= NN * 32) return;
  float di = dinv[i >> 5];
  const float4 v = *reinterpret_cast<const float4*>(agg + (size_t)i * 4);
  float a = v.x * di, b = v.y * di, c = v.z * di, d = v.w * di;
  a = (a > 0.0f) ? a : 0.01f * a;
  b = (b > 0.0f) ? b : 0.01f * b;
  c = (c > 0.0f) ? c : 0.01f * c;
  d = (d > 0.0f) ? d : 0.01f * d;
  uint2 p;
  p.x = (unsigned)f2bf(a) | ((unsigned)f2bf(b) << 16);
  p.y = (unsigned)f2bf(c) | ((unsigned)f2bf(d) << 16);
  *reinterpret_cast<uint2*>(hbf + (size_t)i * 4) = p;
}

// --------------------------- out = h @ W_fc + b_fc   (N x 4)
__global__ void k_fc(const unsigned short* __restrict__ hbf,
                     const float* __restrict__ W,
                     const float* __restrict__ b, float* __restrict__ out) {
  int gid = blockIdx.x * blockDim.x + threadIdx.x;
  if (gid >= NN * NC) return;
  int node = gid >> 2, c = gid & 3;
  float acc = b[c];
  const unsigned short* hr = hbf + (size_t)node * HF;
#pragma unroll 8
  for (int k = 0; k < HF; ++k) acc += bf2f(hr[k]) * W[k * NC + c];
  out[gid] = acc;
}

// ---------------------------------------------------------------- launcher
extern "C" void kernel_launch(void* const* d_in, const int* in_sizes, int n_in,
                              void* d_out, int out_size, void* d_ws, size_t ws_size,
                              hipStream_t stream) {
  const float* x    = (const float*)d_in[0];
  const int*   ei   = (const int*)  d_in[1];
  const float* ea   = (const float*)d_in[2];   // edge_attr (NE x 1)
  const float* W_in = (const float*)d_in[3];
  const float* b_in = (const float*)d_in[4];
  const float* W_h  = (const float*)d_in[5];   // NL x 128 x 128
  const float* b_h  = (const float*)d_in[6];   // NL x 128
  const float* W_fc = (const float*)d_in[7];
  const float* b_fc = (const float*)d_in[8];
  float* out = (float*)d_out;

  // workspace layout (all sub-buffers 16B aligned)
  float* t    = (float*)d_ws;                       // NN*HF f32
  float* agg  = t + (size_t)NN * HF;                // NN*HF f32
  unsigned short* hbf = (unsigned short*)(agg + (size_t)NN * HF); // NN*HF bf16
  float* deg  = (float*)(hbf + (size_t)NN * HF);    // NN f32
  float* dinv = deg + NN;                           // NN f32
  unsigned short* Wbf = (unsigned short*)(dinv + NN); // NL*HF*HF bf16 [l][n][k]

  const int B = 256;
  const int gNH   = (NN * HF + B - 1) / B;
  const int gN    = (NN + B - 1) / B;
  const int gE    = (NE + B - 1) / B;
  const int gScat = (NE * 32 + B - 1) / B;
  const int gPw   = (NN * 32 + B - 1) / B;
  const int gGemm = (NN + 127) / 128;
  const int nW    = NL * HF * HF;

  // degree (constant across layers)
  k_fill0<<<gN, B, 0, stream>>>(deg, NN);
  k_deg  <<<gE, B, 0, stream>>>(ei, deg);
  k_dinv <<<gN, B, 0, stream>>>(deg, dinv);

  // weights -> bf16, transposed to [l][n][k], once
  k_cvt_bf16t<<<(nW + B - 1) / B, B, 0, stream>>>(W_h, Wbf);

  // layer 0: conv(x, W_in, b_in)
  k_gemm_in<<<NN, HF, 0, stream>>>(x, W_in, b_in, t);
  k_fill0  <<<gNH, B, 0, stream>>>(agg, NN * HF);
  k_scatter<<<gScat, B, 0, stream>>>(t, ei, ea, agg);
  k_pointwise<<<gPw, B, 0, stream>>>(agg, dinv, hbf);

  // hidden layers 1..19 (WMMA GEMM)
  for (int l = 0; l < NL; ++l) {
    k_gemm_wmma<<<gGemm, B, 0, stream>>>(hbf, Wbf + (size_t)l * HF * HF,
                                         b_h + (size_t)l * HF, t, NN);
    k_fill0  <<<gNH, B, 0, stream>>>(agg, NN * HF);
    k_scatter<<<gScat, B, 0, stream>>>(t, ei, ea, agg);
    k_pointwise<<<gPw, B, 0, stream>>>(agg, dinv, hbf);
  }

  // final FC
  k_fc<<<(NN * NC + B - 1) / B, B, 0, stream>>>(hbf, W_fc, b_fc, out);
}
